// _MatrixDecomposition2DBase_35493609734400
// MI455X (gfx1250) — matrix operand order repaired and checked by reading, not compiled, not run on hardware
//
#include <hip/hip_runtime.h>

// ---------------------------------------------------------------------------
// NMF (Hamburger MatrixDecomposition2D) on gfx1250.
// B=16, D=512, N=4096, R=64. All matmuls via v_wmma_f32_16x16x32_bf16.
// x converted to bf16 once (67MB, L2-resident); f32 masters kept for
// coef/bases with bf16 shadows used as WMMA operands.
// LDS transposes store packed bf16 pairs (ds_store_b32); big K-loops prefetch
// the next tile (global_prefetch_b8).
// ---------------------------------------------------------------------------

#define BB 16
#define DD 512
#define NN 4096
#define RR 64
#define EPSI 1e-6f
#define INVT 100.0f

typedef __attribute__((ext_vector_type(16))) __bf16 v16bf;
typedef __attribute__((ext_vector_type(8)))  __bf16 v8bf;
typedef __attribute__((ext_vector_type(4)))  __bf16 v4bf;
typedef __attribute__((ext_vector_type(2)))  __bf16 v2bf;
typedef __attribute__((ext_vector_type(8)))  float  v8f;

static __device__ __forceinline__ v16bf frag_cat(v8bf lo, v8bf hi) {
  v16bf r;
#pragma unroll
  for (int i = 0; i < 8; ++i) { r[i] = lo[i]; r[i + 8] = hi[i]; }
  return r;
}

// A fragment (16x32, MxK) from a matrix whose rows are K-contiguous with
// leading dim lda. ISA layout: lane<16 -> row m0+lane, K {0..7,16..23};
// lane>=16 -> row m0+lane-16, K {8..15,24..31}.
static __device__ __forceinline__ v16bf load_frag_a(const __bf16* p, int lda,
                                                    int m0, int k0, int lane) {
  int m  = m0 + (lane & 15);
  int ko = k0 + ((lane >> 4) << 3);
  v8bf lo = *(const v8bf*)(p + m * lda + ko);
  v8bf hi = *(const v8bf*)(p + m * lda + ko + 16);
  return frag_cat(lo, hi);
}

// B fragment (32x16, KxN) from B^T stored row-major (row n = column n of B,
// K contiguous). lane<16 -> col n0+lane, K {0..7,16..23}; lane>=16 -> K {8..15,24..31}.
static __device__ __forceinline__ v16bf load_frag_bt(const __bf16* p, int ldb,
                                                     int n0, int k0, int lane) {
  int n  = n0 + (lane & 15);
  int ko = k0 + ((lane >> 4) << 3);
  v8bf lo = *(const v8bf*)(p + n * ldb + ko);
  v8bf hi = *(const v8bf*)(p + n * ldb + ko + 16);
  return frag_cat(lo, hi);
}

#define WMMA_BF16(a, b, c) \
  __builtin_amdgcn_wmma_f32_16x16x32_bf16(false, (a), false, (b), (short)0, (c), false, false)

// ---------------------------------------------------------------------------
// f32 -> bf16 bulk convert (x)
__global__ void k_cvt_bf16(const float* __restrict__ src,
                           __bf16* __restrict__ dst, int n) {
  int i = (blockIdx.x * blockDim.x + threadIdx.x) * 4;
  if (i + 3 < n) {
    float4 v = *(const float4*)(src + i);
    v4bf o;
    o[0] = (__bf16)v.x; o[1] = (__bf16)v.y; o[2] = (__bf16)v.z; o[3] = (__bf16)v.w;
    *(v4bf*)(dst + i) = o;
  }
}

// copy input bases into f32 master + bf16 shadow (must not mutate d_in)
__global__ void k_init_bases(const float* __restrict__ src,
                             float* __restrict__ dstf,
                             __bf16* __restrict__ dsth, int n) {
  int i = blockIdx.x * blockDim.x + threadIdx.x;
  if (i < n) { float v = src[i]; dstf[i] = v; dsth[i] = (__bf16)v; }
}

// ---------------------------------------------------------------------------
// numc[b][n][r] = sum_d xb[b][d][n] * basesb[b][d][r]     (x^T @ bases)
// Block: 256 thr = 8 waves -> 128x64 tile, K=512 in 32-steps via LDS transpose.
__global__ __launch_bounds__(256) void k_gemm_xtb(const __bf16* __restrict__ xb,
                                                  const __bf16* __restrict__ basesb,
                                                  float* __restrict__ numc) {
  __shared__ __bf16 As[128 * 40];
  __shared__ __bf16 Bs[64 * 40];
  const int b = blockIdx.y;
  const int nbase = blockIdx.x * 128;
  const int t = threadIdx.x, lane = t & 31, wave = t >> 5;
  const __bf16* xB = xb + b * DD * NN;
  const __bf16* bB = basesb + b * DD * RR;
  v8f acc[4] = {};
  for (int kk = 0; kk < DD; kk += 32) {
    {  // x tile: [32 d][128 n] -> As[n][d], packed pairs along d
      int dl = (t >> 4) << 1;        // 0,2,...,30
      int n0 = (t & 15) * 8;         // 0..120
      const __bf16* src = xB + (kk + dl) * NN + nbase + n0;
      v8bf r0 = *(const v8bf*)(src);
      v8bf r1 = *(const v8bf*)(src + NN);
      if (kk + 32 < DD) __builtin_prefetch(src + 32 * NN, 0, 3);
#pragma unroll
      for (int j = 0; j < 8; ++j) {
        v2bf p; p[0] = r0[j]; p[1] = r1[j];
        *(v2bf*)(&As[(n0 + j) * 40 + dl]) = p;
      }
    }
    {  // bases tile: [32 d][64 r] -> Bs[r][d], packed pairs along d
      int dl = (t >> 4) << 1;        // 0,2,...,30
      int r0 = (t & 15) * 4;         // 0..60
      const __bf16* src = bB + (kk + dl) * RR + r0;
      v4bf a0 = *(const v4bf*)(src);
      v4bf a1 = *(const v4bf*)(src + RR);
#pragma unroll
      for (int j = 0; j < 4; ++j) {
        v2bf p; p[0] = a0[j]; p[1] = a1[j];
        *(v2bf*)(&Bs[(r0 + j) * 40 + dl]) = p;
      }
    }
    __syncthreads();
    v16bf a = load_frag_a(As, 40, wave * 16, 0, lane);
#pragma unroll
    for (int tn = 0; tn < 4; ++tn) {
      v16bf bf_ = load_frag_bt(Bs, 40, tn * 16, 0, lane);
      acc[tn] = WMMA_BF16(a, bf_, acc[tn]);
    }
    __syncthreads();
  }
  float* outB = numc + b * NN * RR;
  int mrow = nbase + wave * 16 + ((lane >> 4) << 3);
  int col = lane & 15;
#pragma unroll
  for (int tn = 0; tn < 4; ++tn)
#pragma unroll
    for (int g = 0; g < 8; ++g)
      outB[(mrow + g) * RR + tn * 16 + col] = acc[tn][g];
}

// ---------------------------------------------------------------------------
// Row softmax over R=64: coef = softmax(INVT * numc), + bf16 shadow.
__global__ void k_softmax(const float* __restrict__ S, float* __restrict__ C,
                          __bf16* __restrict__ Ch, int rows) {
  int r = blockIdx.x * blockDim.x + threadIdx.x;
  if (r >= rows) return;
  const float* s = S + (long)r * RR;
  float mx = -1e30f;
  for (int i = 0; i < RR; ++i) mx = fmaxf(mx, s[i]);
  float sum = 0.f;
  for (int i = 0; i < RR; ++i) sum += expf(INVT * (s[i] - mx));
  float inv = 1.f / sum;
  float* c = C + (long)r * RR;
  __bf16* ch = Ch + (long)r * RR;
  for (int i = 0; i < RR; ++i) {
    float v = expf(INVT * (s[i] - mx)) * inv;
    c[i] = v; ch[i] = (__bf16)v;
  }
}

// ---------------------------------------------------------------------------
// Gram: G[b][r][s] = sum_k M[b][k][r] * M[b][k][s]   (M row-major [K][64])
// One block (4 waves) per batch; A and B^T fragments share one LDS transpose.
__global__ __launch_bounds__(128) void k_gram(const __bf16* __restrict__ M, int K,
                                              __bf16* __restrict__ G) {
  __shared__ __bf16 Ts[64 * 40];
  const int b = blockIdx.x;
  const int t = threadIdx.x, lane = t & 31, wave = t >> 5;
  const __bf16* mB = M + (long)b * K * RR;
  v8f acc[4] = {};
  for (int kk = 0; kk < K; kk += 32) {
    // tile [32 k][64 r] -> Ts[r][k], packed pairs along k
    int kl = (t >> 3) << 1;          // 0,2,...,30
    int r0 = (t & 7) * 8;            // 0..56
    const __bf16* src = mB + (kk + kl) * RR + r0;
    v8bf a0 = *(const v8bf*)(src);
    v8bf a1 = *(const v8bf*)(src + RR);
    if (kk + 32 < K) __builtin_prefetch(src + 32 * RR, 0, 3);
#pragma unroll
    for (int j = 0; j < 8; ++j) {
      v2bf p; p[0] = a0[j]; p[1] = a1[j];
      *(v2bf*)(&Ts[(r0 + j) * 40 + kl]) = p;
    }
    __syncthreads();
    v16bf a = load_frag_a(Ts, 40, wave * 16, 0, lane);
#pragma unroll
    for (int tn = 0; tn < 4; ++tn) {
      v16bf bf_ = load_frag_bt(Ts, 40, tn * 16, 0, lane);
      acc[tn] = WMMA_BF16(a, bf_, acc[tn]);
    }
    __syncthreads();
  }
  __bf16* gB = G + b * RR * RR;
  int mrow = wave * 16 + ((lane >> 4) << 3);
  int col = lane & 15;
#pragma unroll
  for (int tn = 0; tn < 4; ++tn)
#pragma unroll
    for (int g = 0; g < 8; ++g)
      gB[(mrow + g) * RR + tn * 16 + col] = (__bf16)acc[tn][g];
}

// ---------------------------------------------------------------------------
// Multiplicative update: P = P * Nu / (Ph @ G + eps); refresh bf16 shadow.
// G is symmetric -> B^T fragments load straight from global; no LDS needed.
__global__ __launch_bounds__(256) void k_mu_update(float* __restrict__ P,
                                                   __bf16* __restrict__ Ph,
                                                   const float* __restrict__ Nu,
                                                   const __bf16* __restrict__ G,
                                                   int rows) {
  const int b = blockIdx.y;
  const int t = threadIdx.x, lane = t & 31, wave = t >> 5;
  const int m0 = blockIdx.x * 128 + wave * 16;
  const __bf16* phB = Ph + (long)b * rows * RR;
  const __bf16* gB = G + b * RR * RR;
  v8f acc[4] = {};
#pragma unroll
  for (int kk = 0; kk < RR; kk += 32) {
    v16bf a = load_frag_a(phB, RR, m0, kk, lane);
#pragma unroll
    for (int tn = 0; tn < 4; ++tn) {
      v16bf bf_ = load_frag_bt(gB, RR, tn * 16, kk, lane);
      acc[tn] = WMMA_BF16(a, bf_, acc[tn]);
    }
  }
  float* pB = P + (long)b * rows * RR;
  const float* nB = Nu + (long)b * rows * RR;
  __bf16* phW = Ph + (long)b * rows * RR;
  int mrow = m0 + ((lane >> 4) << 3);
  int col = lane & 15;
#pragma unroll
  for (int tn = 0; tn < 4; ++tn)
#pragma unroll
    for (int g = 0; g < 8; ++g) {
      int idx = (mrow + g) * RR + tn * 16 + col;
      float v = pB[idx] * nB[idx] / (acc[tn][g] + EPSI);
      pB[idx] = v;
      phW[idx] = (__bf16)v;
    }
}

// ---------------------------------------------------------------------------
// numb[b][d][r] = sum_n xb[b][d][n] * coefb[b][n][r]      (x @ coef)
// A (x) is K-contiguous -> direct global fragments; only B via LDS transpose.
__global__ __launch_bounds__(128) void k_gemm_xc(const __bf16* __restrict__ xb,
                                                 const __bf16* __restrict__ coefb,
                                                 float* __restrict__ numb) {
  __shared__ __bf16 Bs[64 * 40];
  const int b = blockIdx.y;
  const int t = threadIdx.x, lane = t & 31, wave = t >> 5;
  const int m0 = blockIdx.x * 64 + wave * 16;
  const __bf16* xB = xb + b * DD * NN;
  const __bf16* cB = coefb + (long)b * NN * RR;
  v8f acc[4] = {};
  for (int kk = 0; kk < NN; kk += 32) {
    // coef tile [32 n][64 r] -> Bs[r][n], packed pairs along n
    int nl = (t >> 3) << 1;          // 0,2,...,30
    int r0 = (t & 7) * 8;            // 0..56
    const __bf16* src = cB + (kk + nl) * RR + r0;
    v8bf a0 = *(const v8bf*)(src);
    v8bf a1 = *(const v8bf*)(src + RR);
    if (kk + 32 < NN) {
      __builtin_prefetch(src + 32 * RR, 0, 3);                       // next coef tile
      __builtin_prefetch(xB + m0 * NN + kk + 32 + (t & 127) * 16, 0, 3);  // next x frags
    }
#pragma unroll
    for (int j = 0; j < 8; ++j) {
      v2bf p; p[0] = a0[j]; p[1] = a1[j];
      *(v2bf*)(&Bs[(r0 + j) * 40 + nl]) = p;
    }
    __syncthreads();
    v16bf a = load_frag_a(xB, NN, m0, kk, lane);
#pragma unroll
    for (int tn = 0; tn < 4; ++tn) {
      v16bf bf_ = load_frag_bt(Bs, 40, tn * 16, 0, lane);
      acc[tn] = WMMA_BF16(a, bf_, acc[tn]);
    }
    __syncthreads();
  }
  float* outB = numb + b * DD * RR;
  int mrow = m0 + ((lane >> 4) << 3);
  int col = lane & 15;
#pragma unroll
  for (int tn = 0; tn < 4; ++tn)
#pragma unroll
    for (int g = 0; g < 8; ++g)
      outB[(mrow + g) * RR + tn * 16 + col] = acc[tn][g];
}

// ---------------------------------------------------------------------------
// out[b][d][n] = sum_r basesb[d][r] * coefb[n][r]   (bases @ coef^T), K=64.
// Both operands K-contiguous in memory -> LDS-free WMMA kernel.
__global__ __launch_bounds__(256) void k_recon(const __bf16* __restrict__ basesb,
                                               const __bf16* __restrict__ coefb,
                                               float* __restrict__ out) {
  const int b = blockIdx.z;
  const int m0 = blockIdx.x * 16;
  const int t = threadIdx.x, lane = t & 31, wave = t >> 5;
  const int n0 = blockIdx.y * 512 + wave * 64;
  const __bf16* aB = basesb + b * DD * RR;
  const __bf16* cB = coefb + (long)b * NN * RR;
  v8f acc[4] = {};
#pragma unroll
  for (int kk = 0; kk < RR; kk += 32) {
    v16bf a = load_frag_a(aB, RR, m0, kk, lane);
#pragma unroll
    for (int tn = 0; tn < 4; ++tn) {
      v16bf bf_ = load_frag_bt(cB, RR, n0 + tn * 16, kk, lane);
      acc[tn] = WMMA_BF16(a, bf_, acc[tn]);
    }
  }
  float* oB = out + (long)b * DD * NN;
  int mrow = m0 + ((lane >> 4) << 3);
  int col = lane & 15;
#pragma unroll
  for (int tn = 0; tn < 4; ++tn)
#pragma unroll
    for (int g = 0; g < 8; ++g)
      oB[(mrow + g) * NN + n0 + tn * 16 + col] = acc[tn][g];
}

// ---------------------------------------------------------------------------
extern "C" void kernel_launch(void* const* d_in, const int* in_sizes, int n_in,
                              void* d_out, int out_size, void* d_ws, size_t ws_size,
                              hipStream_t stream) {
  const float* x = (const float*)d_in[0];
  const float* bases_in = (const float*)d_in[1];

  char* ws = (char*)d_ws;
  size_t off = 0;
  auto carve = [&](size_t bytes) -> char* {
    char* p = ws + off;
    off += (bytes + 255) & ~(size_t)255;
    return p;
  };
  __bf16* xb     = (__bf16*)carve((size_t)BB * DD * NN * 2);  // 67 MB
  float*  coef   = (float*) carve((size_t)BB * NN * RR * 4);  // 16.8 MB
  __bf16* coefb  = (__bf16*)carve((size_t)BB * NN * RR * 2);  // 8.4 MB
  float*  basesf = (float*) carve((size_t)BB * DD * RR * 4);  // 2.1 MB
  __bf16* basesb = (__bf16*)carve((size_t)BB * DD * RR * 2);  // 1.05 MB
  float*  numc   = (float*) carve((size_t)BB * NN * RR * 4);  // 16.8 MB
  float*  numb   = (float*) carve((size_t)BB * DD * RR * 4);  // 2.1 MB
  __bf16* btbb   = (__bf16*)carve((size_t)BB * RR * RR * 2);
  __bf16* ctcb   = (__bf16*)carve((size_t)BB * RR * RR * 2);
  (void)in_sizes; (void)n_in; (void)out_size; (void)ws_size;

  { int n = BB * DD * NN; k_cvt_bf16<<<n / 4 / 256, 256, 0, stream>>>(x, xb, n); }
  { int n = BB * DD * RR; k_init_bases<<<(n + 255) / 256, 256, 0, stream>>>(bases_in, basesf, basesb, n); }

  dim3 gXTB(NN / 128, BB);
  dim3 gXC(DD / 64, BB);
  dim3 gCU(NN / 128, BB);
  dim3 gBU(DD / 128, BB);

  // coef init: softmax(inv_t * x^T @ bases)
  k_gemm_xtb<<<gXTB, 256, 0, stream>>>(xb, basesb, numc);
  k_softmax<<<(BB * NN) / 256, 256, 0, stream>>>(numc, coef, coefb, BB * NN);

  for (int it = 0; it < 6; ++it) {
    k_gemm_xtb<<<gXTB, 256, 0, stream>>>(xb, basesb, numc);
    k_gram<<<BB, 128, 0, stream>>>(basesb, DD, btbb);
    k_mu_update<<<gCU, 256, 0, stream>>>(coef, coefb, numc, btbb, NN);
    k_gemm_xc<<<gXC, 128, 0, stream>>>(xb, coefb, numb);
    k_gram<<<BB, 128, 0, stream>>>(coefb, NN, ctcb);
    k_mu_update<<<gBU, 256, 0, stream>>>(basesf, basesb, numb, ctcb, DD);
  }

  // final coef update + reconstruction
  k_gemm_xtb<<<gXTB, 256, 0, stream>>>(xb, basesb, numc);
  k_gram<<<BB, 128, 0, stream>>>(basesb, DD, btbb);
  k_mu_update<<<gCU, 256, 0, stream>>>(coef, coefb, numc, btbb, NN);
  k_recon<<<dim3(DD / 16, NN / 512, BB), 256, 0, stream>>>(basesb, coefb, (float*)d_out);
}
